// MultiHeadedAttention_v2_22617297781017
// MI455X (gfx1250) — matrix operand order repaired and checked by reading, not compiled, not run on hardware
//
#include <hip/hip_runtime.h>

// ---------------------------------------------------------------------------
// MultiHeadedAttention (channel-token attention) + BatchNorm for MI455X
// bf16 WMMA (v_wmma_f32_16x16x32_bf16) GEMM core, fp32 accumulate.
// gfx1250 async-to-LDS tile staging; native v_cvt bf16 conversion.
// B=32, C=256, HW=4096, NH=4, D=64, ND=256
// ---------------------------------------------------------------------------

typedef __attribute__((ext_vector_type(16))) __bf16 v16bf;
typedef __attribute__((ext_vector_type(8)))  float  v8f;
typedef int v4i_vs __attribute__((vector_size(16)));
typedef __attribute__((address_space(1))) v4i_vs* gptr_v4i;
typedef __attribute__((address_space(3))) v4i_vs* lptr_v4i;

#define WGSIZE 256
#define BM 128
#define BK 32
#define AS (BK + 8)   // padded LDS row stride (elements) -> 80B, 16B-aligned rows
#define BSS (BK + 8)

#if __has_builtin(__builtin_amdgcn_global_load_async_to_lds_b128) && \
    __has_builtin(__builtin_amdgcn_s_wait_asynccnt)
#define USE_ASYNC 1
#else
#define USE_ASYNC 0
#endif

// fp32 -> bf16 via native hardware convert (RNE); lowers to v_cvt_*bf16*
__device__ __forceinline__ unsigned short f2bf(float f) {
  __bf16 h = (__bf16)f;
  return __builtin_bit_cast(unsigned short, h);
}

// 16-byte global->LDS copy: async DMA path on gfx1250, vector copy fallback
__device__ __forceinline__ void copy16_g2l(const unsigned short* g, unsigned short* l) {
#if USE_ASYNC
  __builtin_amdgcn_global_load_async_to_lds_b128((gptr_v4i)g, (lptr_v4i)l, 0, 0);
#else
  *(uint4*)l = *(const uint4*)g;
#endif
}

__device__ __forceinline__ void async_join() {
#if USE_ASYNC
  __builtin_amdgcn_s_wait_asynccnt(0);
#endif
}

union FragU {
  v16bf v;
  uint4 u[2];
};

// Generic batched GEMM: C[z] = alpha * A[z] (MxK) * B[z] (KxN)
//   BNT:      N block-tile (128 for big GEMMs, 64 for O=P*V)
//   A_F32:    A is fp32 in global, converted to bf16 while staging to LDS
//   B_TRANS:  B stored row-major [N,K] bf16 (K-contiguous; enables async staging)
//   OUT_BF16: store bf16, else fp32
// grid = (N/BNT, M/BM, nz);  z decomposed as z0 = z/nz1, z1 = z%nz1
template<int BNT, int A_F32, int B_TRANS, int OUT_BF16>
__global__ __launch_bounds__(WGSIZE)
void gemm_wmma(const void* __restrict__ Ap, const unsigned short* __restrict__ Bp,
               void* __restrict__ Cp,
               int K, int lda, int ldb, int ldc,
               long long sA0, long long sA1,
               long long sB0, long long sB1,
               long long sC0, long long sC1,
               int nz1, float alpha)
{
  constexpr int NT = BNT / 32;            // 16-wide N tiles per wave (2 or 4)
  __shared__ __align__(16) unsigned short Asm[BM * AS];
  __shared__ __align__(16) unsigned short Bsm[BNT * BSS];

  const int tid = threadIdx.x;
  const int z   = blockIdx.z;
  const int z0  = z / nz1;
  const int z1  = z - z0 * nz1;

  const long long aoff = (long long)z0 * sA0 + (long long)z1 * sA1;
  const float*          Af = (const float*)Ap + aoff;
  const unsigned short* Ab = (const unsigned short*)Ap + aoff;
  const unsigned short* B  = Bp + (long long)z0 * sB0 + (long long)z1 * sB1;

  const int m0 = blockIdx.y * BM;
  const int n0 = blockIdx.x * BNT;

  const int lane = tid & 31;
  const int wave = tid >> 5;
  const int wm   = wave >> 1;        // 0..3  (M sub-tile of 32 rows)
  const int wn   = wave & 1;         // 0..1  (N sub-tile of BNT/2 cols)
  const int half = lane >> 4;        // lane group
  const int l15  = lane & 15;

  v8f acc[2][NT];
  for (int mt = 0; mt < 2; ++mt)
    for (int nt = 0; nt < NT; ++nt)
      for (int j = 0; j < 8; ++j) acc[mt][nt][j] = 0.0f;

  for (int k0 = 0; k0 < K; k0 += BK) {
    __syncthreads();
    // ---- stage A tile [BM x BK] ----
    if (A_F32) {
      const float* Abase = Af + (size_t)m0 * lda + k0;
      for (int i = tid; i < (BM * BK) / 4; i += WGSIZE) {
        int r  = i >> 3;           // BK/4 == 8 chunks/row
        int c4 = (i & 7) << 2;
        float4 f = *(const float4*)(Abase + (size_t)r * lda + c4);
        uint2 pk;
        pk.x = (unsigned int)f2bf(f.x) | ((unsigned int)f2bf(f.y) << 16);
        pk.y = (unsigned int)f2bf(f.z) | ((unsigned int)f2bf(f.w) << 16);
        *(uint2*)&Asm[r * AS + c4] = pk;
      }
    } else {
      const unsigned short* Abase = Ab + (size_t)m0 * lda + k0;
      for (int i = tid; i < (BM * BK) / 8; i += WGSIZE) {
        int r  = i >> 2;           // BK/8 == 4 chunks/row
        int c8 = (i & 3) << 3;
        copy16_g2l(Abase + (size_t)r * lda + c8, &Asm[r * AS + c8]);
      }
    }
    // ---- stage B tile, stored K-contiguous per output column: Bsm[n][k] ----
    if (B_TRANS) {
      const unsigned short* Bbase = B + (size_t)n0 * ldb + k0;
      for (int i = tid; i < (BNT * BK) / 8; i += WGSIZE) {
        int n  = i >> 2;
        int c8 = (i & 3) << 3;
        copy16_g2l(Bbase + (size_t)n * ldb + c8, &Bsm[n * BSS + c8]);
      }
    } else {
      const unsigned short* Bbase = B + (size_t)k0 * ldb + n0;
      for (int i = tid; i < BNT * BK; i += WGSIZE) {
        int k = i / BNT;
        int n = i % BNT;           // coalesced in n
        Bsm[n * BSS + k] = Bbase[(size_t)k * ldb + n];
      }
    }
    async_join();
    __syncthreads();

    // ---- build fragments per ISA 7.12.2 layouts and issue WMMA ----
    const int ka0 = half ? 8  : 0;   // A elems 0..7  hold K = ka0..ka0+7
    const int ka1 = half ? 24 : 16;  // A elems 8..15 hold K = ka1..ka1+7
    const int kb0 = half ? 8 : 0;    // B elems 0..7 hold K = kb0..kb0+7, elems 8..15 hold K = kb0+16..kb0+23

    FragU a[2], b[NT];
    for (int mt = 0; mt < 2; ++mt) {
      int r = wm * 32 + mt * 16 + l15;
      a[mt].u[0] = *(const uint4*)&Asm[r * AS + ka0];
      a[mt].u[1] = *(const uint4*)&Asm[r * AS + ka1];
    }
    for (int nt = 0; nt < NT; ++nt) {
      int n = wn * (BNT / 2) + nt * 16 + l15;
      b[nt].u[0] = *(const uint4*)&Bsm[n * BSS + kb0];
      b[nt].u[1] = *(const uint4*)&Bsm[n * BSS + kb0 + 16];
    }
    for (int mt = 0; mt < 2; ++mt)
      for (int nt = 0; nt < NT; ++nt)
        acc[mt][nt] = __builtin_amdgcn_wmma_f32_16x16x32_bf16(
            false, a[mt].v, false, b[nt].v, (short)0, acc[mt][nt], false, false);
  }

  // ---- epilogue: C element (M = 8*half + j, N = l15) per 16x16 tile ----
  const long long coff = (long long)z0 * sC0 + (long long)z1 * sC1;
  for (int mt = 0; mt < 2; ++mt) {
    for (int nt = 0; nt < NT; ++nt) {
      int rb = m0 + wm * 32 + mt * 16 + half * 8;
      int c  = n0 + wn * (BNT / 2) + nt * 16 + l15;
      for (int j = 0; j < 8; ++j) {
        float v = alpha * acc[mt][nt][j];
        size_t idx = (size_t)coff + (size_t)(rb + j) * ldc + c;
        if (OUT_BF16) ((unsigned short*)Cp)[idx] = f2bf(v);
        else          ((float*)Cp)[idx] = v;
      }
    }
  }
}

// fp32 [K,N] -> bf16 [N,K] transposing cast (LDS tiled, coalesced both sides)
__global__ __launch_bounds__(256)
void cast_transpose_bf16(const float* __restrict__ src, unsigned short* __restrict__ dst,
                         int K, int N) {
  __shared__ float t[32][33];
  const int kb = blockIdx.x * 32, nb = blockIdx.y * 32;
  const int tx = threadIdx.x & 31, ty = threadIdx.x >> 5;   // 8 rows of 32
  for (int j = ty; j < 32; j += 8) t[j][tx] = src[(size_t)(kb + j) * N + nb + tx];
  __syncthreads();
  for (int j = ty; j < 32; j += 8) dst[(size_t)(nb + j) * K + kb + tx] = f2bf(t[tx][j]);
}

// row softmax over 256-wide rows; wave per row; bf16 output
__global__ __launch_bounds__(256)
void softmax_rows(const float* __restrict__ S, unsigned short* __restrict__ P) {
  const int wave = threadIdx.x >> 5;
  const int lane = threadIdx.x & 31;
  const size_t row = (size_t)blockIdx.x * 8 + wave;
  const float* sr = S + row * 256;
  float v[8];
  float mx = -3.4e38f;
  #pragma unroll
  for (int i = 0; i < 8; ++i) { v[i] = sr[lane + i * 32]; mx = fmaxf(mx, v[i]); }
  #pragma unroll
  for (int off = 16; off; off >>= 1) mx = fmaxf(mx, __shfl_xor(mx, off, 32));
  float s = 0.0f;
  #pragma unroll
  for (int i = 0; i < 8; ++i) { v[i] = __expf(v[i] - mx); s += v[i]; }
  #pragma unroll
  for (int off = 16; off; off >>= 1) s += __shfl_xor(s, off, 32);
  const float inv = 1.0f / s;
  unsigned short* pr = P + row * 256;
  #pragma unroll
  for (int i = 0; i < 8; ++i) pr[lane + i * 32] = f2bf(v[i] * inv);
}

// per-channel mean / rstd over [B=32, C=256, HW=4096]
__global__ __launch_bounds__(256)
void chan_stats(const float* __restrict__ out, float* __restrict__ mean, float* __restrict__ rstd) {
  const int c = blockIdx.x;
  const int tid = threadIdx.x;
  float s = 0.0f, s2 = 0.0f;
  for (int b = 0; b < 32; ++b) {
    const float* p = out + ((size_t)b * 256 + c) * 4096;
    for (int i = tid; i < 4096; i += 256) { float v = p[i]; s += v; s2 += v * v; }
  }
  #pragma unroll
  for (int off = 16; off; off >>= 1) { s += __shfl_xor(s, off, 32); s2 += __shfl_xor(s2, off, 32); }
  __shared__ float sh[16];
  const int wave = tid >> 5, lane = tid & 31;
  if (lane == 0) { sh[wave] = s; sh[8 + wave] = s2; }
  __syncthreads();
  if (tid == 0) {
    float S = 0.0f, S2 = 0.0f;
    for (int w = 0; w < 8; ++w) { S += sh[w]; S2 += sh[8 + w]; }
    const float n = 32.0f * 4096.0f;
    float m = S / n;
    float var = S2 / n - m * m;
    mean[c] = m;
    rstd[c] = rsqrtf(var + 1e-5f);
  }
}

// in-place BN: out = (out-mean[c])*rstd[c]*gamma[c] + beta[c]
__global__ __launch_bounds__(256)
void bn_apply(float* __restrict__ out,
              const float* __restrict__ mean, const float* __restrict__ rstd,
              const float* __restrict__ gamma, const float* __restrict__ beta) {
  size_t i = (size_t)blockIdx.x * blockDim.x + threadIdx.x;
  int c = (int)((i >> 12) & 255);
  out[i] = (out[i] - mean[c]) * rstd[c] * gamma[c] + beta[c];
}

// ---------------------------------------------------------------------------

extern "C" void kernel_launch(void* const* d_in, const int* in_sizes, int n_in,
                              void* d_out, int out_size, void* d_ws, size_t ws_size,
                              hipStream_t stream) {
  (void)in_sizes; (void)n_in; (void)out_size; (void)ws_size;
  const float* x     = (const float*)d_in[0];
  const float* x1    = (const float*)d_in[1];
  const float* Wq    = (const float*)d_in[2];
  const float* Wk    = (const float*)d_in[3];
  const float* Wv    = (const float*)d_in[4];
  const float* Wfc   = (const float*)d_in[5];
  const float* gamma = (const float*)d_in[6];
  const float* beta  = (const float*)d_in[7];
  float* out = (float*)d_out;

  char* ws = (char*)d_ws;
  unsigned short* Wqt  = (unsigned short*)(ws + 0);            // bf16 [256,4096], 2 MiB each
  unsigned short* Wkt  = (unsigned short*)(ws + (1ull << 21));
  unsigned short* Wvt  = (unsigned short*)(ws + (2ull << 21));
  unsigned short* Wfct = (unsigned short*)(ws + (3ull << 21)); // bf16 [4096,256]
  unsigned short* qb   = (unsigned short*)(ws + (4ull << 21)); // bf16 [b,256,256], 4 MiB each
  unsigned short* kb   = (unsigned short*)(ws + (6ull << 21));
  unsigned short* vb   = (unsigned short*)(ws + (8ull << 21));
  unsigned short* ctx  = (unsigned short*)(ws + (10ull << 21));
  unsigned short* P    = (unsigned short*)(ws + (12ull << 21)); // bf16 [b,h,256,256], 16 MiB
  float*          S    = (float*)(ws + (20ull << 21));          // fp32, 32 MiB
  float*          meanp = (float*)(ws + (36ull << 21));
  float*          rstdp = (float*)(ws + (36ull << 21) + 4096);

  const dim3 blk(WGSIZE);

  // transposed bf16 weight casts:  W[K,N] -> Wt[N,K]
  cast_transpose_bf16<<<dim3(4096 / 32, 256 / 32), 256, 0, stream>>>(Wq,  Wqt,  4096, 256);
  cast_transpose_bf16<<<dim3(4096 / 32, 256 / 32), 256, 0, stream>>>(Wk,  Wkt,  4096, 256);
  cast_transpose_bf16<<<dim3(4096 / 32, 256 / 32), 256, 0, stream>>>(Wv,  Wvt,  4096, 256);
  cast_transpose_bf16<<<dim3(256 / 32, 4096 / 32), 256, 0, stream>>>(Wfc, Wfct, 256, 4096);

  // Q = (x @ Wq) / 8 ; K,V from x1.  M=256 N=256 K=4096, batched over b=32
  gemm_wmma<128, 1, 1, 1><<<dim3(2, 2, 32), blk, 0, stream>>>(
      x,  Wqt, qb, 4096, 4096, 4096, 256,
      1048576LL, 0LL, 0LL, 0LL, 65536LL, 0LL, 1, 0.125f);
  gemm_wmma<128, 1, 1, 1><<<dim3(2, 2, 32), blk, 0, stream>>>(
      x1, Wkt, kb, 4096, 4096, 4096, 256,
      1048576LL, 0LL, 0LL, 0LL, 65536LL, 0LL, 1, 1.0f);
  gemm_wmma<128, 1, 1, 1><<<dim3(2, 2, 32), blk, 0, stream>>>(
      x1, Wvt, vb, 4096, 4096, 4096, 256,
      1048576LL, 0LL, 0LL, 0LL, 65536LL, 0LL, 1, 1.0f);

  // S[b,h] = Q_h @ K_h^T : M=256 N=256 K=64, z = b*4+h (nz1=4)
  gemm_wmma<128, 0, 1, 0><<<dim3(2, 2, 128), blk, 0, stream>>>(
      qb, kb, S, 64, 256, 256, 256,
      65536LL, 64LL, 65536LL, 64LL, 262144LL, 65536LL, 4, 1.0f);

  // softmax over last axis; 32768 rows, 8 rows/block
  softmax_rows<<<4096, 256, 0, stream>>>(S, P);

  // ctx[b,:,h*64:] = P[b,h] @ V_h : M=256 N=64 K=256  (B row-major [K,N])
  gemm_wmma<64, 0, 0, 1><<<dim3(1, 2, 128), blk, 0, stream>>>(
      P, vb, ctx, 256, 256, 256, 256,
      262144LL, 65536LL, 65536LL, 64LL, 65536LL, 64LL, 4, 1.0f);

  // out = ctx @ Wfc : M=256 N=4096 K=256, batched over b=32, fp32 out
  gemm_wmma<128, 0, 1, 0><<<dim3(32, 2, 32), blk, 0, stream>>>(
      ctx, Wfct, out, 256, 256, 256, 4096,
      65536LL, 0LL, 0LL, 0LL, 1048576LL, 0LL, 1, 1.0f);

  // BatchNorm (batch statistics)
  chan_stats<<<256, 256, 0, stream>>>(out, meanp, rstdp);
  bn_apply<<<(32u * 256u * 4096u) / 256u, 256, 0, stream>>>(out, meanp, rstdp, gamma, beta);
}
